// MultiHeadAttention_19121194402332
// MI455X (gfx1250) — matrix operand order repaired and checked by reading, not compiled, not run on hardware
//
#include <hip/hip_runtime.h>
#include <cstdint>

// ---------------------------------------------------------------------------
// MultiHeadAttention forward for MI455X (gfx1250, wave32, WMMA 16x16x32 f16)
//
// Pipeline (all on `stream`):
//  1) cvt_f32_f16 : q,k,v and Wq/Wk/Wv/Wo -> f16 staging in d_ws
//  2) gemm_nt     : qh = q@Wq^T+b (head-split), kh likewise,
//                   vhT = (v@Wv^T+b) stored TRANSPOSED per head [64, L]
//  3) attn_fused  : per (head, 16-row strip):
//                     scores (WMMA) -> LDS (128KB strip, stride-padded)
//                     softmax NORMALIZED in LDS
//                     raw_att -> HBM once, via GLOBAL_STORE_ASYNC_FROM_LDS_B128
//                       (ASYNCcnt), fully overlapped with the att@V WMMA loop
//                     O = att@V (A from LDS, B from vhT), stored directly in
//                       the reference's (buggy) merged ctx[B,L,D] layout
//  4) gemm_nt     : P = ctx@Wo^T + bo  (f32)
//  5) add_layernorm: out = LN(P + q)*gamma + beta
// ---------------------------------------------------------------------------

typedef __attribute__((ext_vector_type(16))) _Float16 v16h;
typedef __attribute__((ext_vector_type(8)))  _Float16 v8h;
typedef __attribute__((ext_vector_type(8)))  float    v8f;
typedef __attribute__((__vector_size__(4 * sizeof(int)))) int vsi4;

constexpr int kB  = 4;      // batch
constexpr int kL  = 2048;   // sequence length
constexpr int kD  = 512;    // model dim
constexpr int kH  = 8;      // heads
constexpr int kDh = 64;     // head dim
constexpr float kScale = 0.125f;   // 1/sqrt(64)
constexpr float kEps   = 1e-5f;

// LDS strip: 16 rows x kL f32, row stride padded so rows r and r+8 land on
// different bank groups (2052*8 mod 64 == 32) and rows stay 16B aligned.
constexpr int kSld = 2052;
constexpr int kSmemScores = 16 * kSld * 4;            // 131,328 B
constexpr int kSmemPart   = 8 * 256 * 4;              //   8,192 B
constexpr int kSmemRed    = 256 * 4;                  //   1,024 B
constexpr int kSmemTotal  = kSmemScores + kSmemPart + kSmemRed;  // 140,544 B

// ---- CDNA5 async LDS->global store path (ASYNCcnt), guarded ---------------
#ifdef __has_builtin
#if __has_builtin(__builtin_amdgcn_global_store_async_from_lds_b128) && \
    __has_builtin(__builtin_amdgcn_s_wait_asynccnt)
#define USE_ASYNC_STORE 1
#endif
#endif
#ifndef USE_ASYNC_STORE
#define USE_ASYNC_STORE 0
#endif

// Builtin expects int4-typed pointers: addrspace(1) global, addrspace(3) LDS.
#define AS1V(p) ((__attribute__((address_space(1))) vsi4*)(uintptr_t)(p))
#define AS3V(p) ((__attribute__((address_space(3))) vsi4*)(uintptr_t)(p))

// ---------------- WMMA fragment helpers (ISA 7.12.2 layouts) ---------------

// A fragment, 16(M) x 32(K) f16, source row-major with leading dim `ld`.
// lane&15 = M row; halves 0..7 = K (lane>>4)*8 .. +7 ; halves 8..15 = +16.
__device__ __forceinline__ v16h frag_ld_a_h(const _Float16* p, int ld) {
  const int lane = threadIdx.x & 31;
  const int r  = lane & 15;
  const int kk = (lane >> 4) << 3;
  const _Float16* q = p + (size_t)r * ld + kk;
  v8h lo = *(const v8h*)(q);
  v8h hi = *(const v8h*)(q + 16);
  v16h f;
#pragma unroll
  for (int i = 0; i < 8; ++i) { f[i] = lo[i]; f[i + 8] = hi[i]; }
  return f;
}

// Same A fragment but converting from f32 source (generic ptr; used on LDS).
__device__ __forceinline__ v16h frag_ld_a_f32(const float* p, int ld) {
  const int lane = threadIdx.x & 31;
  const int r  = lane & 15;
  const int kk = (lane >> 4) << 3;
  const float* q = p + (size_t)r * ld + kk;
  v16h f;
#pragma unroll
  for (int i = 0; i < 8; ++i) f[i]     = (_Float16)q[i];
#pragma unroll
  for (int i = 0; i < 8; ++i) f[i + 8] = (_Float16)q[i + 16];
  return f;
}

// B fragment, 32(K) x 16(N) f16, sourced from the NT operand stored as
// N x K row-major. lane&15 = N column; halves 0..7 = K (lane>>4)*8 .. +7 ;
// halves 8..15 = +16.
__device__ __forceinline__ v16h frag_ld_b_h(const _Float16* p, int ld) {
  const int lane = threadIdx.x & 31;
  const int n  = lane & 15;
  const int kk = (lane >> 4) << 3;
  const _Float16* q = p + (size_t)n * ld + kk;
  v8h lo = *(const v8h*)(q);
  v8h hi = *(const v8h*)(q + 16);
  v16h f;
#pragma unroll
  for (int i = 0; i < 8; ++i) { f[i] = lo[i]; f[i + 8] = hi[i]; }
  return f;
}

// C/D 16x16 f32 tile store: VGPR i -> M = i + (lane>=16 ? 8 : 0), N = lane&15.
__device__ __forceinline__ void frag_st_f(float* p, int ld, v8f c) {
  const int lane = threadIdx.x & 31;
  const int n  = lane & 15;
  const int mb = (lane >> 4) << 3;
#pragma unroll
  for (int i = 0; i < 8; ++i) p[(size_t)(i + mb) * ld + n] = c[i];
}

// ---------------------------- kernels --------------------------------------

__global__ void cvt_f32_f16(const float* __restrict__ in,
                            _Float16* __restrict__ out, int n) {
  int i = blockIdx.x * blockDim.x + threadIdx.x;
  if (i < n) out[i] = (_Float16)in[i];
}

// Generic NT GEMM: Y[m,n] = sum_k X[m,k]*W[n,k] + bias[n].
// M = kB*kL = 8192, N = K = kD = 512.
// Block = 256 threads (8 waves as 2x4), block tile 32(M) x 64(N).
// mode 0: store f16 head-split  outH[(b*H+h)*L*dh + l*dh + dcol]
// mode 1: store f16 head-split TRANSPOSED outH[(b*H+h)*dh*L + dcol*L + l]
// mode 2: store f32 row-major   outF[m*kD + n]
__global__ __launch_bounds__(256)
void gemm_nt(const _Float16* __restrict__ X, const _Float16* __restrict__ W,
             const float* __restrict__ bias, _Float16* __restrict__ outH,
             float* __restrict__ outF, int mode) {
  const int wave = threadIdx.x >> 5;
  const int m0 = blockIdx.x * 32 + (wave >> 2) * 16;
  const int n0 = blockIdx.y * 64 + (wave & 3) * 16;

  const _Float16* a = X + (size_t)m0 * kD;
  const _Float16* b = W + (size_t)n0 * kD;

  v8f c = {};
#pragma unroll 4
  for (int k = 0; k < kD; k += 32) {
    __builtin_prefetch(a + k + 256, 0, 1);     // global_prefetch_b8
    __builtin_prefetch(b + k + 256, 0, 1);
    v16h af = frag_ld_a_h(a + k, kD);
    v16h bf = frag_ld_b_h(b + k, kD);
    c = __builtin_amdgcn_wmma_f32_16x16x32_f16(false, af, false, bf,
                                               (short)0, c, false, false);
  }

  const int lane = threadIdx.x & 31;
  const int n  = lane & 15;
  const int mb = (lane >> 4) << 3;
  const float bv = bias[n0 + n];
#pragma unroll
  for (int i = 0; i < 8; ++i) c[i] += bv;

  if (mode == 2) {
    frag_st_f(outF + (size_t)m0 * kD + n0, kD, c);
  } else {
    const int ncol = n0 + n;
    const int h = ncol >> 6;          // head
    const int dcol = ncol & 63;       // dim within head
#pragma unroll
    for (int i = 0; i < 8; ++i) {
      const int m = m0 + mb + i;
      const int bi = m >> 11;         // batch  (m / kL)
      const int l  = m & 2047;        // position
      const size_t base = (size_t)(bi * kH + h) * (size_t)(kL * kDh);
      const size_t dst = (mode == 0) ? base + (size_t)l * kDh + dcol
                                     : base + (size_t)dcol * kL + l;
      outH[dst] = (_Float16)c[i];
    }
  }
}

// Fused attention: one workgroup per (16-query strip, head).
// grid = (kL/16, kB*kH), block = 256 (8 wave32 waves), dyn LDS = kSmemTotal.
__global__ __launch_bounds__(256)
void attn_fused(const _Float16* __restrict__ qh, const _Float16* __restrict__ kh,
                const _Float16* __restrict__ vhT, float* __restrict__ att,
                _Float16* __restrict__ ctx) {
  extern __shared__ char smem[];
  float* S    = (float*)smem;                              // [16][kSld]
  float* part = (float*)(smem + kSmemScores);              // [8][16][16]
  float* red  = (float*)(smem + kSmemScores + kSmemPart);  // [256]

  const int head = blockIdx.y;          // bh = b*kH + h
  const int m0   = blockIdx.x * 16;
  const int t    = threadIdx.x;
  const int wave = t >> 5;
  const int lane = t & 31;
  const int ln   = lane & 15;
  const int lmb  = (lane >> 4) << 3;

  // ---- Q fragments for this strip: 16 rows x 64 (two 16x32 frags), once ----
  const _Float16* aq = qh + (size_t)head * kL * kDh + (size_t)m0 * kDh;
  v16h qa0 = frag_ld_a_h(aq, kDh);
  v16h qa1 = frag_ld_a_h(aq + 32, kDh);

  // ---- scores: each wave covers 16 column tiles of 16 -> LDS (scaled) ----
  const _Float16* kb = kh + (size_t)head * kL * kDh;
  for (int it = 0; it < 16; ++it) {
    const int n0 = it * 128 + wave * 16;
    const _Float16* b = kb + (size_t)n0 * kDh;
    __builtin_prefetch(b + 128 * kDh, 0, 1);   // next column tile
    v16h b0 = frag_ld_b_h(b, kDh);
    v16h b1 = frag_ld_b_h(b + 32, kDh);
    v8f c = {};
    c = __builtin_amdgcn_wmma_f32_16x16x32_f16(false, qa0, false, b0,
                                               (short)0, c, false, false);
    c = __builtin_amdgcn_wmma_f32_16x16x32_f16(false, qa1, false, b1,
                                               (short)0, c, false, false);
#pragma unroll
    for (int i = 0; i < 8; ++i)
      S[(i + lmb) * kSld + n0 + ln] = c[i] * kScale;
  }
  __syncthreads();

  // ---- softmax over 16 rows x 2048, 16 threads per row; normalize in LDS --
  const int r  = t >> 4;        // row 0..15
  const int cs = t & 15;        // chunk within row
  float* row = S + r * kSld;

  float mx = -3.402823e38f;
#pragma unroll 8
  for (int i = 0; i < 128; ++i) mx = fmaxf(mx, row[cs + 16 * i]);
  red[t] = mx; __syncthreads();
  for (int s = 8; s > 0; s >>= 1) {
    if (cs < s) red[t] = fmaxf(red[t], red[t + s]);
    __syncthreads();
  }
  mx = red[r * 16];
  __syncthreads();

  float sum = 0.f;
#pragma unroll 8
  for (int i = 0; i < 128; ++i) {
    const float e = __expf(row[cs + 16 * i] - mx);
    row[cs + 16 * i] = e;
    sum += e;
  }
  red[t] = sum; __syncthreads();
  for (int s = 8; s > 0; s >>= 1) {
    if (cs < s) red[t] += red[t + s];
    __syncthreads();
  }
  const float inv = 1.0f / red[r * 16];
#pragma unroll 8
  for (int i = 0; i < 128; ++i) row[cs + 16 * i] *= inv;   // final softmax
  __syncthreads();

  // ---- raw_att writeback: LDS -> HBM exactly once.
  // Async DMA (ASYNCcnt) when available: the strip is read-only for the rest
  // of the kernel, so the stores drain overlapped with the att@V WMMA loop.
  {
    float* ag = att + (size_t)head * kL * kL;   // per-head [L, L]
    for (int e4 = t; e4 < 16 * (kL / 4); e4 += 256) {
      const int rr = e4 >> 9;          // / (kL/4)
      const int c4 = e4 & 511;
      float*       gp = ag + (size_t)(m0 + rr) * kL + c4 * 4;
      const float* lp = S + rr * kSld + c4 * 4;
#if USE_ASYNC_STORE
      __builtin_amdgcn_global_store_async_from_lds_b128(AS1V(gp), AS3V(lp),
                                                        0, 0);
#else
      *(float4*)gp = *(const float4*)lp;
#endif
    }
  }

  // ---- O = att @ V : 8 waves = 4 N-tiles x 2 K-halves, A from LDS ----
  const int nt    = wave & 3;
  const int khalf = wave >> 2;
  const _Float16* bv = vhT + (size_t)head * kDh * kL
                           + (size_t)(nt * 16) * kL + khalf * 1024;
  const float* arow = S + khalf * 1024;
  v8f c = {};
  for (int k = 0; k < 1024; k += 32) {
    __builtin_prefetch(bv + k + 128, 0, 1);
    v16h af = frag_ld_a_f32(arow + k, kSld);
    v16h bf = frag_ld_b_h(bv + k, kL);
    c = __builtin_amdgcn_wmma_f32_16x16x32_f16(false, af, false, bf,
                                               (short)0, c, false, false);
  }
#pragma unroll
  for (int i = 0; i < 8; ++i)
    part[wave * 256 + (i + lmb) * 16 + ln] = c[i];
  __syncthreads();

  // Reduce the two K-halves and store O directly in the reference's (buggy)
  // merged layout: flat (B*H,L,dh) read as (H,B,L,dh) -> ctx[b2, l, h2*64+d],
  // with h2 = head/kB, b2 = head%kB.
  const int h2 = head >> 2;   // head / kB
  const int b2 = head & 3;    // head % kB
  _Float16* og = ctx + ((size_t)b2 * kL + m0) * kD + h2 * kDh;
  for (int e = t; e < 16 * kDh; e += 256) {
    const int rr = e >> 6;
    const int cg = e & 63;
    const int n2 = cg >> 4;
    const int cc = cg & 15;
    const float val =
        part[n2 * 256 + rr * 16 + cc] + part[(n2 + 4) * 256 + rr * 16 + cc];
    og[(size_t)rr * kD + cg] = (_Float16)val;
  }

#if USE_ASYNC_STORE
  __builtin_amdgcn_s_wait_asynccnt(0);   // drain raw_att DMA before endpgm
#endif
}

// x = P + residual; LayerNorm over D=512; write to output.
__global__ __launch_bounds__(256)
void add_layernorm(const float* __restrict__ P, const float* __restrict__ resid,
                   const float* __restrict__ gamma, const float* __restrict__ beta,
                   float* __restrict__ out) {
  const int row = blockIdx.x;
  const int t = threadIdx.x;
  const float* pp = P + (size_t)row * kD;
  const float* rr = resid + (size_t)row * kD;

  float x0 = pp[t] + rr[t];
  float x1 = pp[t + 256] + rr[t + 256];

  __shared__ float red[256];
  red[t] = x0 + x1; __syncthreads();
  for (int s = 128; s > 0; s >>= 1) {
    if (t < s) red[t] += red[t + s];
    __syncthreads();
  }
  const float mu = red[0] * (1.0f / kD); __syncthreads();

  const float d0 = x0 - mu, d1 = x1 - mu;
  red[t] = d0 * d0 + d1 * d1; __syncthreads();
  for (int s = 128; s > 0; s >>= 1) {
    if (t < s) red[t] += red[t + s];
    __syncthreads();
  }
  const float rstd = rsqrtf(red[0] * (1.0f / kD) + kEps);

  float* o = out + (size_t)row * kD;
  o[t]       = d0 * rstd * gamma[t]       + beta[t];
  o[t + 256] = d1 * rstd * gamma[t + 256] + beta[t + 256];
}

// ---------------------------- launcher -------------------------------------

extern "C" void kernel_launch(void* const* d_in, const int* in_sizes, int n_in,
                              void* d_out, int out_size, void* d_ws, size_t ws_size,
                              hipStream_t stream) {
  const float* q    = (const float*)d_in[0];
  const float* k    = (const float*)d_in[1];
  const float* v    = (const float*)d_in[2];
  const float* Wq   = (const float*)d_in[3];
  const float* bq   = (const float*)d_in[4];
  const float* Wk   = (const float*)d_in[5];
  const float* bk   = (const float*)d_in[6];
  const float* Wv   = (const float*)d_in[7];
  const float* bv   = (const float*)d_in[8];
  const float* Wo   = (const float*)d_in[9];
  const float* bo   = (const float*)d_in[10];
  const float* gmm  = (const float*)d_in[11];
  const float* beta = (const float*)d_in[12];

  float* out_ln = (float*)d_out;                              // [B, L, D]
  float* att    = out_ln + (size_t)kB * kL * kD;              // [B*H, L, L]

  // workspace carve-out (256B aligned)
  char* w = (char*)d_ws;
  size_t off = 0;
  auto alloc = [&](size_t bytes) -> void* {
    off = (off + 255) & ~(size_t)255;
    void* p = w + off;
    off += bytes;
    return p;
  };
  const size_t nX = (size_t)kB * kL * kD;   // 4,194,304
  const size_t nW = (size_t)kD * kD;        // 262,144

  _Float16* xq16 = (_Float16*)alloc(nX * 2);
  _Float16* xk16 = (_Float16*)alloc(nX * 2);
  _Float16* xv16 = (_Float16*)alloc(nX * 2);
  _Float16* Wq16 = (_Float16*)alloc(nW * 2);
  _Float16* Wk16 = (_Float16*)alloc(nW * 2);
  _Float16* Wv16 = (_Float16*)alloc(nW * 2);
  _Float16* Wo16 = (_Float16*)alloc(nW * 2);
  _Float16* qh   = (_Float16*)alloc(nX * 2);   // [B*H, L, dh]
  _Float16* kh   = (_Float16*)alloc(nX * 2);   // [B*H, L, dh]
  _Float16* vhT  = (_Float16*)alloc(nX * 2);   // [B*H, dh, L]
  _Float16* ctx  = (_Float16*)alloc(nX * 2);   // [B, L, D] (merged)
  float*    P    = (float*)   alloc(nX * 4);   // [B*L, D]

  // 1) f32 -> f16 staging
  cvt_f32_f16<<<(int)((nX + 255) / 256), 256, 0, stream>>>(q, xq16, (int)nX);
  cvt_f32_f16<<<(int)((nX + 255) / 256), 256, 0, stream>>>(k, xk16, (int)nX);
  cvt_f32_f16<<<(int)((nX + 255) / 256), 256, 0, stream>>>(v, xv16, (int)nX);
  cvt_f32_f16<<<(int)((nW + 255) / 256), 256, 0, stream>>>(Wq, Wq16, (int)nW);
  cvt_f32_f16<<<(int)((nW + 255) / 256), 256, 0, stream>>>(Wk, Wk16, (int)nW);
  cvt_f32_f16<<<(int)((nW + 255) / 256), 256, 0, stream>>>(Wv, Wv16, (int)nW);
  cvt_f32_f16<<<(int)((nW + 255) / 256), 256, 0, stream>>>(Wo, Wo16, (int)nW);

  // 2) projections (head-split; V transposed per head)
  dim3 gp(kB * kL / 32, kD / 64);                      // (256, 8)
  gemm_nt<<<gp, 256, 0, stream>>>(xq16, Wq16, bq, qh, nullptr, 0);
  gemm_nt<<<gp, 256, 0, stream>>>(xk16, Wk16, bk, kh, nullptr, 0);
  gemm_nt<<<gp, 256, 0, stream>>>(xv16, Wv16, bv, vhT, nullptr, 1);

  // 3) fused scores + softmax + att@V + merge (raw_att hits HBM exactly once)
  dim3 ga(kL / 16, kB * kH);                           // (128, 32)
  attn_fused<<<ga, 256, kSmemTotal, stream>>>(qh, kh, vhT, att, ctx);

  // 4) output projection (f32 row-major result)
  gemm_nt<<<gp, 256, 0, stream>>>(ctx, Wo16, bo, nullptr, P, 2);

  // 5) residual + LayerNorm -> out
  add_layernorm<<<kB * kL, 256, 0, stream>>>(P, q, gmm, beta, out_ln);
}
